// LambdaAttention_89051851915773
// MI455X (gfx1250) — matrix operand order repaired and checked by reading, not compiled, not run on hardware
//
#include <hip/hip_runtime.h>

typedef __bf16 bf16;
typedef __attribute__((ext_vector_type(16))) __bf16 v16bf;
typedef __attribute__((ext_vector_type(8)))  float  v8f;

static constexpr int BB = 16, S = 1024, D = 512, H = 8, Dk = 16, Dv = 64;
static constexpr int MTOK = BB * S;      // 16384 tokens
static constexpr int NP   = 256;         // padded projection width
static constexpr float EPS = 1e-5f;

union VPun { v16bf v; uint4 u[2]; };

#define WMMA_BF16(A, Bv, C) \
    __builtin_amdgcn_wmma_f32_16x16x32_bf16(false, (A), false, (Bv), (short)0, (C), false, false)

// CDNA5 async global->LDS copy (ASYNCcnt path). lds_off = low 32 bits of generic LDS address.
__device__ __forceinline__ void async_g2l_b128(unsigned lds_off, const void* gaddr) {
    asm volatile("global_load_async_to_lds_b128 %0, %1, off"
                 :: "v"(lds_off), "v"(gaddr) : "memory");
}
__device__ __forceinline__ void wait_async0() {
    asm volatile("s_wait_asynccnt 0" ::: "memory");
}
__device__ __forceinline__ unsigned lds_off32(const void* p) {
    return (unsigned)(unsigned long long)p;   // LDS aperture: addr[31:0] is the LDS byte offset
}

// ---------------------------------------------------------------- converts
__global__ void cvt_f32_bf16(const float* __restrict__ in, bf16* __restrict__ out, int n) {
    int i = blockIdx.x * 256 + threadIdx.x;
    if (i < n) out[i] = (bf16)in[i];
}

// Wcat rows: [0,128)=Wq, [128,144)=Wk, [144,208)=Wv, [208,256)=0.  [256][512], K contiguous
__global__ void build_wcat(const float* __restrict__ Wq, const float* __restrict__ Wk,
                           const float* __restrict__ Wv, bf16* __restrict__ Wcat) {
    int i = blockIdx.x * 256 + threadIdx.x;
    int row = i >> 9, col = i & 511;
    float v = 0.f;
    if (row < 128)      v = Wq[row * 512 + col];
    else if (row < 144) v = Wk[(row - 128) * 512 + col];
    else if (row < 208) v = Wv[(row - 144) * 512 + col];
    Wcat[i] = (bf16)v;
}

// pos_embed [n][m][k] f32 -> ET [n][k][m] bf16 (LDS tile transpose, padded)
__global__ void transpose_pos(const float* __restrict__ pos, bf16* __restrict__ ET) {
    __shared__ float tile[128 * 17];
    int n = blockIdx.x, m0 = blockIdx.y * 128;
    const float* src = pos + (size_t)n * S * Dk + (size_t)m0 * Dk;
#pragma unroll
    for (int j = 0; j < 8; ++j) {
        int idx = threadIdx.x + j * 256;
        tile[(idx >> 4) * 17 + (idx & 15)] = src[idx];
    }
    __syncthreads();
    bf16* dst = ET + (size_t)n * Dk * S + m0;
#pragma unroll
    for (int j = 0; j < 8; ++j) {
        int idx = threadIdx.x + j * 256;
        int k = idx >> 7, m = idx & 127;
        dst[(size_t)k * S + m] = (bf16)tile[m * 17 + k];
    }
}

// ------------------------------------------------- generic bf16 WMMA GEMM (global operands)
// C[M][ldc](f32) = A[M][K] @ Bw[N][K]^T. wave tile 16x64; block = 4 waves; K % 64 == 0.
__global__ void __launch_bounds__(128)
wmma_gemm_bf16(const bf16* __restrict__ A, const bf16* __restrict__ Bw,
               float* __restrict__ C, int K, int ldc) {
    int wave = threadIdx.x >> 5;
    int lane = threadIdx.x & 31;
    int half = lane >> 4, l15 = lane & 15;
    int mrow  = blockIdx.x * 16;
    int ntile = (blockIdx.y * 4 + wave) * 64;

    const bf16* aBase = A  + (size_t)(mrow + l15) * K + half * 8;
    const bf16* bBase = Bw + (size_t)(ntile + l15) * K + half * 8;

    v8f acc[4] = {};
    VPun a0, a1, b0[4], b1[4];

    auto loadA = [&](VPun& a, int kb) {
        a.u[0] = *(const uint4*)(aBase + kb);
        a.u[1] = *(const uint4*)(aBase + kb + 16);
    };
    auto loadB = [&](VPun* bt, int kb) {
#pragma unroll
        for (int t = 0; t < 4; ++t) {
            const bf16* bp = bBase + (size_t)t * 16 * K + kb;
            bt[t].u[0] = *(const uint4*)(bp);
            bt[t].u[1] = *(const uint4*)(bp + 16);
        }
    };

    loadA(a0, 0); loadB(b0, 0);
    for (int kb = 0; kb < K; kb += 64) {
        loadA(a1, kb + 32); loadB(b1, kb + 32);               // prefetch stage 1
#pragma unroll
        for (int t = 0; t < 4; ++t) acc[t] = WMMA_BF16(a0.v, b0[t].v, acc[t]);
        if (kb + 64 < K) { loadA(a0, kb + 64); loadB(b0, kb + 64); }  // prefetch next stage 0
#pragma unroll
        for (int t = 0; t < 4; ++t) acc[t] = WMMA_BF16(a1.v, b1[t].v, acc[t]);
    }
#pragma unroll
    for (int t = 0; t < 4; ++t)
#pragma unroll
        for (int r = 0; r < 8; ++r)
            C[(size_t)(mrow + r + half * 8) * ldc + (ntile + t * 16 + l15)] = acc[t][r];
}

// ---------------------------------------------------------------- BN stats
__global__ void bn_stats(const float* __restrict__ proj, float* __restrict__ mean,
                         float* __restrict__ rstd) {
    int c = blockIdx.x;
    int col = (c < 128) ? c : (144 + (c - 128));
    __shared__ float s1[256], s2[256];
    float a = 0.f, b = 0.f;
    for (int i = threadIdx.x; i < MTOK; i += 256) {
        float v = proj[(size_t)i * NP + col];
        a += v; b += v * v;
    }
    s1[threadIdx.x] = a; s2[threadIdx.x] = b; __syncthreads();
    for (int off = 128; off > 0; off >>= 1) {
        if (threadIdx.x < off) { s1[threadIdx.x] += s1[threadIdx.x + off];
                                 s2[threadIdx.x] += s2[threadIdx.x + off]; }
        __syncthreads();
    }
    if (threadIdx.x == 0) {
        float m = s1[0] / MTOK;
        mean[c] = m; rstd[c] = rsqrtf(s2[0] / MTOK - m * m + EPS);
    }
}

// ---------------------------------------------------------------- softmax over S per (b, ch)
__global__ void k_softmax(const float* __restrict__ proj, float* __restrict__ ksoft) {
    int b = blockIdx.x >> 4, ch = blockIdx.x & 15;
    __shared__ float red[256];
    size_t base = (size_t)b * S;
    float mx = -1e30f;
    for (int s = threadIdx.x; s < S; s += 256)
        mx = fmaxf(mx, proj[(base + s) * NP + 128 + ch]);
    red[threadIdx.x] = mx; __syncthreads();
    for (int off = 128; off > 0; off >>= 1) {
        if (threadIdx.x < off) red[threadIdx.x] = fmaxf(red[threadIdx.x], red[threadIdx.x + off]);
        __syncthreads();
    }
    mx = red[0]; __syncthreads();
    float sum = 0.f;
    for (int s = threadIdx.x; s < S; s += 256)
        sum += __expf(proj[(base + s) * NP + 128 + ch] - mx);
    red[threadIdx.x] = sum; __syncthreads();
    for (int off = 128; off > 0; off >>= 1) {
        if (threadIdx.x < off) red[threadIdx.x] += red[threadIdx.x + off];
        __syncthreads();
    }
    float inv = 1.0f / red[0];
    for (int s = threadIdx.x; s < S; s += 256)
        ksoft[(base + s) * 16 + ch] = __expf(proj[(base + s) * NP + 128 + ch] - mx) * inv;
}

// ---------------------------------------------------------------- BN apply
__global__ void apply_bn_q(const float* __restrict__ proj, const float* __restrict__ mean,
                           const float* __restrict__ rstd, const float* __restrict__ g,
                           const float* __restrict__ be, float* __restrict__ qbn) {
    int i = blockIdx.x * 256 + threadIdx.x;
    int row = i >> 7, c = i & 127;
    qbn[i] = (proj[(size_t)row * NP + c] - mean[c]) * rstd[c] * g[c] + be[c];
}

__global__ void apply_bn_v_t(const float* __restrict__ proj, const float* __restrict__ mean,
                             const float* __restrict__ rstd, const float* __restrict__ g,
                             const float* __restrict__ be, bf16* __restrict__ VT) {
    int i = blockIdx.x * 256 + threadIdx.x;      // < 16*64*1024 ; VT[b][v][m]
    int m = i & 1023, v = (i >> 10) & 63, b = i >> 16;
    float x = proj[((size_t)(b * 1024 + m)) * NP + 144 + v];
    VT[i] = (bf16)((x - mean[128 + v]) * rstd[128 + v] * g[v] + be[v]);
}

// ---------------------------------------------------------------- content lambda (tiny)
__global__ void content_lambda_k(const float* __restrict__ ksoft, const bf16* __restrict__ VT,
                                 float* __restrict__ clam) {
    int b = blockIdx.x, t = threadIdx.x;         // k = t/64, v = t%64
    int k = t >> 6, v = t & 63;
    const float* kp = ksoft + (size_t)b * S * 16 + k;
    const bf16* vp = VT + (size_t)b * 64 * S + (size_t)v * S;
    float acc = 0.f;
    for (int m = 0; m < S; ++m) acc += kp[m * 16] * (float)vp[m];
    clam[b * 1024 + t] = acc;
}

// ---------------------------------------------------------------- THE big kernel
// grid = (S/16, BB); block = 256 thr = 8 waves. b = blockIdx.y shared by block;
// wave w handles n0 = blockIdx.x*16 + 2w and n0+1 (register-blocked: 2 A streams
// share each B register tile -> halves LDS bandwidth per WMMA).
// VT[b] staged into LDS in 32KB chunks, double-buffered via async global->LDS
// (ASYNCcnt) + workgroup barrier. A (ET) streams from global (LOADcnt), ping-pong.
static constexpr int MCH  = 256;                 // m-chunk width
static constexpr int LDST = MCH + 8;             // padded LDS row stride (bf16): +4 banks/row
__global__ void __launch_bounds__(256)
lambda_pos(const bf16* __restrict__ ET, const bf16* __restrict__ VT,
           const float* __restrict__ qbn, const float* __restrict__ clam,
           bf16* __restrict__ Yb) {
    __shared__ bf16 ldsB[2][64 * LDST];          // 2 x 33 KB

    int b = blockIdx.y;
    int wave = threadIdx.x >> 5;
    int n0 = blockIdx.x * 16 + wave * 2;         // this wave's two n-tiles
    int lane = threadIdx.x & 31;
    int half = lane >> 4, l15 = lane & 15;

    const bf16* VTb    = VT + (size_t)b * 64 * S;
    const bf16* aBase0 = ET + (size_t)n0 * Dk * S + (size_t)l15 * S + half * 8;
    const bf16* aBase1 = aBase0 + (size_t)Dk * S;

    auto stage = [&](int buf, int mc) {          // copy VT[b][:, mc..mc+256) -> ldsB[buf]
        const bf16* src = VTb + mc;
#pragma unroll
        for (int i = 0; i < 8; ++i) {
            int g = threadIdx.x + i * 256;       // 2048 x 16B granules
            int row = g >> 5;
            int off = (g & 31) * 8;
            async_g2l_b128(lds_off32(&ldsB[buf][row * LDST + off]),
                           src + (size_t)row * S + off);
        }
    };
    auto loadA2 = [&](VPun* a, int m) {          // both n-tiles' A fragments
        a[0].u[0] = *(const uint4*)(aBase0 + m);
        a[0].u[1] = *(const uint4*)(aBase0 + m + 16);
        a[1].u[0] = *(const uint4*)(aBase1 + m);
        a[1].u[1] = *(const uint4*)(aBase1 + m + 16);
    };
    auto loadB = [&](VPun* bt, int buf, int km) {
#pragma unroll
        for (int t = 0; t < 4; ++t) {
            const bf16* p = &ldsB[buf][(t * 16 + l15) * LDST + km + half * 8];
            bt[t].u[0] = *(const uint4*)(p);
            bt[t].u[1] = *(const uint4*)(p + 16);
        }
    };

    v8f acc[2][4] = {};
    stage(0, 0);
    wait_async0();
    __syncthreads();

    for (int c = 0; c < S / MCH; ++c) {
        int mc = c * MCH;
        int buf = c & 1;
        if (c + 1 < S / MCH) stage(buf ^ 1, mc + MCH);   // async prefetch next chunk

        VPun a0[2], a1[2], b0[4], b1[4];
        loadA2(a0, mc); loadB(b0, buf, 0);
#pragma unroll
        for (int kk = 0; kk < MCH; kk += 64) {
            loadA2(a1, mc + kk + 32); loadB(b1, buf, kk + 32);
#pragma unroll
            for (int t = 0; t < 4; ++t) acc[0][t] = WMMA_BF16(a0[0].v, b0[t].v, acc[0][t]);
#pragma unroll
            for (int t = 0; t < 4; ++t) acc[1][t] = WMMA_BF16(a0[1].v, b0[t].v, acc[1][t]);
            if (kk + 64 < MCH) { loadA2(a0, mc + kk + 64); loadB(b0, buf, kk + 64); }
#pragma unroll
            for (int t = 0; t < 4; ++t) acc[0][t] = WMMA_BF16(a1[0].v, b1[t].v, acc[0][t]);
#pragma unroll
            for (int t = 0; t < 4; ++t) acc[1][t] = WMMA_BF16(a1[1].v, b1[t].v, acc[1][t]);
        }
        if (c + 1 < S / MCH) { wait_async0(); __syncthreads(); }
    }

    // epilogue per n-tile: fuse content lambda, contract with q, store Y (bf16)
    const float* cl = clam + b * 1024;
#pragma unroll
    for (int j = 0; j < 2; ++j) {
        int n = n0 + j;
#pragma unroll
        for (int t = 0; t < 4; ++t)
#pragma unroll
            for (int r = 0; r < 8; ++r)
                acc[j][t][r] += cl[(r + half * 8) * 64 + (t * 16 + l15)];

        const float* qp = qbn + ((size_t)(b * 1024 + n)) * 128 + half * 8;
        bf16* yrow = Yb + ((size_t)(b * 1024 + n)) * 512;
        for (int h = 0; h < H; ++h) {
            float q[8];
#pragma unroll
            for (int r = 0; r < 8; ++r) q[r] = qp[h * 16 + r];
#pragma unroll
            for (int t = 0; t < 4; ++t) {
                float p = 0.f;
#pragma unroll
                for (int r = 0; r < 8; ++r) p += q[r] * acc[j][t][r];
                p += __shfl_xor(p, 16, 32);
                if (half == 0) yrow[h * 64 + t * 16 + l15] = (bf16)p;
            }
        }
    }
}

// ---------------------------------------------------------------- launcher
extern "C" void kernel_launch(void* const* d_in, const int* in_sizes, int n_in,
                              void* d_out, int out_size, void* d_ws, size_t ws_size,
                              hipStream_t stream) {
    const float* x    = (const float*)d_in[0];
    const float* Wq   = (const float*)d_in[1];
    const float* Wk   = (const float*)d_in[2];
    const float* Wv   = (const float*)d_in[3];
    const float* gq   = (const float*)d_in[4];
    const float* bq   = (const float*)d_in[5];
    const float* gv   = (const float*)d_in[6];
    const float* bv   = (const float*)d_in[7];
    const float* pos  = (const float*)d_in[8];
    const float* Wout = (const float*)d_in[9];
    float* out = (float*)d_out;

    char* w = (char*)d_ws;
    size_t o = 0;
    auto take = [&](size_t bytes) { char* p = w + o; o += (bytes + 255) & ~(size_t)255; return p; };
    bf16*  ET    = (bf16*)take((size_t)S * Dk * S * 2);
    bf16*  xb    = (bf16*)take((size_t)MTOK * D * 2);
    bf16*  Wcat  = (bf16*)take((size_t)NP * D * 2);
    bf16*  Woutb = (bf16*)take((size_t)D * (H * Dv) * 2);
    float* proj  = (float*)take((size_t)MTOK * NP * 4);
    float* mean  = (float*)take(192 * 4);
    float* rstd  = (float*)take(192 * 4);
    float* ksoft = (float*)take((size_t)BB * S * 16 * 4);
    float* qbn   = (float*)take((size_t)MTOK * 128 * 4);
    bf16*  VT    = (bf16*)take((size_t)BB * 64 * S * 2);
    float* clam  = (float*)take((size_t)BB * 16 * 64 * 4);
    bf16*  Yb    = (bf16*)take((size_t)MTOK * 512 * 2);

    cvt_f32_bf16<<<(MTOK * D + 255) / 256, 256, 0, stream>>>(x, xb, MTOK * D);
    build_wcat<<<(NP * D) / 256, 256, 0, stream>>>(Wq, Wk, Wv, Wcat);
    cvt_f32_bf16<<<(D * H * Dv + 255) / 256, 256, 0, stream>>>(Wout, Woutb, D * H * Dv);
    transpose_pos<<<dim3(S, S / 128), 256, 0, stream>>>(pos, ET);

    wmma_gemm_bf16<<<dim3(MTOK / 16, 1), 128, 0, stream>>>(xb, Wcat, proj, D, NP);

    bn_stats<<<192, 256, 0, stream>>>(proj, mean, rstd);
    k_softmax<<<BB * 16, 256, 0, stream>>>(proj, ksoft);
    apply_bn_q<<<(MTOK * 128) / 256, 256, 0, stream>>>(proj, mean, rstd, gq, bq, qbn);
    apply_bn_v_t<<<(BB * 64 * S) / 256, 256, 0, stream>>>(proj, mean, rstd, gv, bv, VT);

    content_lambda_k<<<BB, 1024, 0, stream>>>(ksoft, VT, clam);

    lambda_pos<<<dim3(S / 16, BB), 256, 0, stream>>>(ET, VT, qbn, clam, Yb);

    wmma_gemm_bf16<<<dim3(MTOK / 16, 2), 128, 0, stream>>>(Yb, Woutb, out, D, 512);
}
